// ProteinGNN_78632261255700
// MI455X (gfx1250) — matrix operand order repaired and checked by reading, not compiled, not run on hardware
//
#include <hip/hip_runtime.h>
#include <hip/hip_bf16.h>
#include <hip/hip_fp16.h>
#include <math.h>

// ---------------------------------------------------------------------------
// ProteinGNN on MI455X (gfx1250): WMMA GEMMs (packed B fragments, vector A
// loads, no guards) + atomic scatter-softmax GATv2 edge kernels.
// ---------------------------------------------------------------------------

#define EPSV 1e-5f
#define NEGS 0.2f

typedef __attribute__((ext_vector_type(16))) _Float16 v16h;
typedef __attribute__((ext_vector_type(8)))  _Float16 v8h;
typedef __attribute__((ext_vector_type(8)))  float    v8f;

// ------------------------- device helpers ----------------------------------

__device__ __forceinline__ void atomicMaxF(float* addr, float v) {
  // IEEE trick: signed-int max for v>=0, unsigned min for v<0.
  if (v >= 0.f) atomicMax((int*)addr, __float_as_int(v));
  else          atomicMin((unsigned int*)addr, (unsigned int)__float_as_int(v));
}

__device__ __forceinline__ void load_a_run(const float* p, v16h& a, int j0) {
  v8f r = *(const v8f*)p;            // 32B contiguous -> 2x b128
#pragma unroll
  for (int j = 0; j < 8; ++j) a[j0 + j] = (_Float16)r[j];
}
__device__ __forceinline__ void load_a_run(const _Float16* p, v16h& a, int j0) {
  v8h r = *(const v8h*)p;            // 16B contiguous -> b128
#pragma unroll
  for (int j = 0; j < 8; ++j) a[j0 + j] = r[j];
}

// ------------------------- WMMA GEMM ---------------------------------------
// Wave w of each 128-thread block computes one 16x16 tile of
// C = A[M,Kp] * W[Kp,N] + bias.  A: row-major f32 or f16, lda = Kp (padded,
// multiple of 32, no bound checks).  W: pre-packed per-lane v16h fragments:
// Wf[((ntile*nKs + ks)*32 + lane)] with element j of the fragment = W(k,n),
// k = ks*32 + (lane>>4)*16 + j, n = ntile*16 + (lane&15).
template <typename TA>
__global__ void __launch_bounds__(128)
gemm_wmma(const TA* __restrict__ A, int lda,
          const v16h* __restrict__ Wf, int nKs,
          const float* __restrict__ bias,
          float* __restrict__ C, int ldc, int M)
{
  const int lane = threadIdx.x & 31;
  const int wid  = threadIdx.x >> 5;
  const int row0 = (blockIdx.y * 4 + wid) << 4;
  if (row0 >= M) return;
  const int col0 = blockIdx.x << 4;

  const int mA    = row0 + (lane & 15);
  const int kOffA = (lane >> 4) * 16;  // A frag: lanes 0-15 cover K 0..15, lanes 16-31 K 16..31
  const TA* Arow  = A + (size_t)mA * lda + kOffA;
  const v16h* Wp  = Wf + (size_t)blockIdx.x * nKs * 32 + lane;

  v8f acc = {};
  for (int ks = 0; ks < nKs; ++ks) {
    v16h a, b;
    b = Wp[ks * 32];
    load_a_run(Arow + ks * 32,      a, 0);
    load_a_run(Arow + ks * 32 + 8,  a, 8);
    acc = __builtin_amdgcn_wmma_f32_16x16x32_f16(false, a, false, b,
                                                 (short)0, acc, false, false);
  }

  const int mC = row0 + (lane >> 4) * 8;
  const int nC = col0 + (lane & 15);
  const float bv = bias[nC];
#pragma unroll
  for (int r = 0; r < 8; ++r)
    C[(size_t)(mC + r) * ldc + nC] = acc[r] + bv;
}

// ------------------------- pack / pad kernels ------------------------------

// Pack f32 W[K,N] (row-major) into WMMA B fragments, zero-padded to Kp=nKs*32.
__global__ void packB_kernel(const float* __restrict__ W, _Float16* __restrict__ Wf,
                             int K, int N, int nKs) {
  int idx = blockIdx.x * blockDim.x + threadIdx.x;
  int total = (N >> 4) * nKs * 512;       // == N * Kp halves
  if (idx >= total) return;
  int j    = idx & 15;
  int lane = (idx >> 4) & 31;
  int ks   = (idx >> 9) % nKs;
  int nt   = idx / (nKs << 9);
  int k = ks * 32 + ((lane >> 4) << 4) + j;
  int n = (nt << 4) + (lane & 15);
  Wf[idx] = (k < K) ? (_Float16)W[(size_t)k * N + n] : (_Float16)0.f;
}

// Pad/convert f32 X[Nr,K] -> f16 X[Nr,Kp], zero-padded columns.
__global__ void padA_kernel(const float* __restrict__ X, _Float16* __restrict__ Xh,
                            int Nr, int K, int Kp) {
  int i = blockIdx.x * blockDim.x + threadIdx.x;
  if (i >= Nr * Kp) return;
  int r = i / Kp, k = i - r * Kp;
  Xh[i] = (k < K) ? (_Float16)X[(size_t)r * K + k] : (_Float16)0.f;
}

// ------------------------- fused PReLU + BN --------------------------------

__global__ void bn_stats_prelu_kernel(const float* __restrict__ x,
                                      const float* __restrict__ aP, int aScalar,
                                      float* __restrict__ sums, int Nr, int C) {
  int c = threadIdx.x;                 // blockDim.x == C, coalesced row sweeps
  float av = aP[aScalar ? 0 : c];
  float s = 0.f, s2 = 0.f;
  for (int r = blockIdx.x; r < Nr; r += gridDim.x) {
    float v = x[(size_t)r * C + c];
    v = (v >= 0.f) ? v : av * v;
    s += v; s2 += v * v;
  }
  atomicAdd(&sums[c], s);
  atomicAdd(&sums[C + c], s2);
}

template <typename TO>
__global__ void bn_apply_prelu_kernel(const float* __restrict__ x,
                                      const float* __restrict__ aP, int aScalar,
                                      const float* __restrict__ sums,
                                      const float* __restrict__ g, const float* __restrict__ b,
                                      TO* __restrict__ out, int Nr, int C) {
  int i = blockIdx.x * blockDim.x + threadIdx.x;
  if (i >= Nr * C) return;
  int c = i % C;
  float v = x[i];
  float av = aP[aScalar ? 0 : c];
  v = (v >= 0.f) ? v : av * v;
  float inv = 1.f / (float)Nr;
  float mu  = sums[c] * inv;
  float var = sums[C + c] * inv - mu * mu;
  out[i] = (TO)((v - mu) * rsqrtf(var + EPSV) * g[c] + b[c]);
}

// ------------------------- misc elementwise --------------------------------

__global__ void fill_kernel(float* p, float v, int n) {
  int i = blockIdx.x * blockDim.x + threadIdx.x;
  if (i < n) p[i] = v;
}

__global__ void add2bias_kernel(float* x, const float* __restrict__ b1,
                                const float* __restrict__ b2, int n, int C) {
  int i = blockIdx.x * blockDim.x + threadIdx.x;
  if (i >= n) return;
  int c = i % C;
  x[i] += b1[c] + b2[c];
}

__global__ void concat_kernel(const float* __restrict__ hpr, const float* __restrict__ g2,
                              float* __restrict__ z, int B, int H) {
  int i = blockIdx.x * blockDim.x + threadIdx.x;
  int C2 = 2 * H;
  if (i >= B * C2) return;
  int r = i / C2, c = i % C2;
  z[i] = (c < H) ? hpr[(size_t)r * H + c] : g2[(size_t)r * H + (c - H)];
}

// ------------------------- GATv2 edge kernels ------------------------------
// One wave per edge; 32 lanes cover H=128 columns (4 each).

__global__ void edge_score_kernel(const int* __restrict__ src, const int* __restrict__ dst,
                                  const float* __restrict__ xl, const float* __restrict__ xr,
                                  const float* __restrict__ att,
                                  float* __restrict__ esc, float* __restrict__ emax, int E) {
  int e = blockIdx.x * (blockDim.x >> 5) + (threadIdx.x >> 5);
  if (e >= E) return;
  int lane = threadIdx.x & 31;
  int s = src[e], d = dst[e];
  const float* pl = xl + (size_t)s * 128;
  const float* pr = xr + (size_t)d * 128;
  float part = 0.f;
#pragma unroll
  for (int c = lane; c < 128; c += 32) {
    float m = pl[c] + pr[c];
    m = (m >= 0.f) ? m : NEGS * m;
    part += m * att[c];
  }
#pragma unroll
  for (int w = 16; w > 0; w >>= 1) part += __shfl_xor(part, w);
  if (lane == 0) {
    esc[e] = part;
    atomicMaxF(emax + d, part);
  }
}

__global__ void edge_exp_kernel(const int* __restrict__ dst, float* __restrict__ esc,
                                const float* __restrict__ emax, float* __restrict__ denom,
                                int E) {
  int e = blockIdx.x * blockDim.x + threadIdx.x;
  if (e >= E) return;
  int d = dst[e];
  float ex = __expf(esc[e] - emax[d]);
  esc[e] = ex;
  atomicAdd(denom + d, ex);
}

__global__ void edge_scatter_kernel(const int* __restrict__ src, const int* __restrict__ dst,
                                    const float* __restrict__ esc, const float* __restrict__ denom,
                                    const float* __restrict__ xl, float* __restrict__ out, int E) {
  int e = blockIdx.x * (blockDim.x >> 5) + (threadIdx.x >> 5);
  if (e >= E) return;
  int lane = threadIdx.x & 31;
  int s = src[e], d = dst[e];
  float alpha = esc[e] / denom[d];
  const float* pl = xl + (size_t)s * 128;
  float* po = out + (size_t)d * 128;
#pragma unroll
  for (int c = lane; c < 128; c += 32)
    atomicAdd(po + c, alpha * pl[c]);
}

// ------------------------- host orchestration ------------------------------

static inline int imin_(int a, int b) { return a < b ? a : b; }

struct GatP { const float *Wl, *bl, *Wr, *br, *att, *bias; };

extern "C" void kernel_launch(void* const* d_in, const int* in_sizes, int n_in,
                              void* d_out, int out_size, void* d_ws, size_t ws_size,
                              hipStream_t stream) {
  (void)in_sizes; (void)n_in; (void)out_size; (void)ws_size;

  const int N_AA = 500000, N_PROT = 10000, F_AA = 25, F_PROT = 480;
  const int H = 128, V = 4096, B = 2048;
  const int E_BP = 500000, E_PP = 320000;

  // --- input pytree (setup_inputs dict order, params flattened depth-first)
  const float* x_aa     = (const float*)d_in[0];
  const float* x_prot   = (const float*)d_in[1];
  const float* Win_aa   = (const float*)d_in[2];
  const float* bin_aa   = (const float*)d_in[3];
  const float* Win_prot = (const float*)d_in[4];
  const float* bin_prot = (const float*)d_in[5];
  const float* a1_aa    = (const float*)d_in[6];
  const float* a1_prot  = (const float*)d_in[7];
  const float* g1_aa    = (const float*)d_in[8];
  const float* b1_aa    = (const float*)d_in[9];
  const float* g1_prot  = (const float*)d_in[10];
  const float* b1_prot  = (const float*)d_in[11];
  GatP c1bp = {(const float*)d_in[12], (const float*)d_in[13], (const float*)d_in[14],
               (const float*)d_in[15], (const float*)d_in[16], (const float*)d_in[17]};
  GatP c1pp = {(const float*)d_in[18], (const float*)d_in[19], (const float*)d_in[20],
               (const float*)d_in[21], (const float*)d_in[22], (const float*)d_in[23]};
  const float* a_g1 = (const float*)d_in[24];
  const float* g_g1 = (const float*)d_in[25];
  const float* b_g1 = (const float*)d_in[26];
  GatP c2bp = {(const float*)d_in[27], (const float*)d_in[28], (const float*)d_in[29],
               (const float*)d_in[30], (const float*)d_in[31], (const float*)d_in[32]};
  GatP c2pp = {(const float*)d_in[33], (const float*)d_in[34], (const float*)d_in[35],
               (const float*)d_in[36], (const float*)d_in[37], (const float*)d_in[38]};
  const float* a_g2   = (const float*)d_in[39];
  const float* g_g2   = (const float*)d_in[40];
  const float* b_g2   = (const float*)d_in[41];
  const float* a_pr   = (const float*)d_in[42];
  const float* g_pr   = (const float*)d_in[43];
  const float* b_pr   = (const float*)d_in[44];
  const float* Wpost  = (const float*)d_in[45];
  const float* bpost  = (const float*)d_in[46];
  const float* a_post = (const float*)d_in[47];
  const float* g_post = (const float*)d_in[48];
  const float* b_post = (const float*)d_in[49];
  const float* Wout   = (const float*)d_in[50];
  const float* bout   = (const float*)d_in[51];
  const int* ei_bp    = (const int*)d_in[52];   // [2, E_BP]: row0=src(aa), row1=dst(prot)
  const int* ei_pp    = (const int*)d_in[53];   // [2, E_PP]

  // --- workspace bump allocator (256B aligned)
  char* base = (char*)d_ws;
  size_t off = 0;
  auto alloc = [&](size_t bytes) -> void* {
    void* p = base + off;
    off += (bytes + 255) & ~(size_t)255;
    return p;
  };

  float*     yaa    = (float*)alloc((size_t)N_AA * H * 4);   // GEMM out; reused as xl_big
  _Float16*  h_aa   = (_Float16*)alloc((size_t)N_AA * H * 2);
  _Float16*  xa16   = (_Float16*)alloc((size_t)N_AA * 32 * 2);  // padded x_aa
  float*     h_pr   = (float*)alloc((size_t)N_PROT * H * 4);
  float*     smallA = (float*)alloc((size_t)N_PROT * H * 4);
  float*     smallB = (float*)alloc((size_t)N_PROT * H * 4);
  float*     g1     = (float*)alloc((size_t)N_PROT * H * 4);
  float*     g2     = (float*)alloc((size_t)N_PROT * H * 4);
  float*     z      = (float*)alloc((size_t)B * 2 * H * 4);
  float*     z2     = (float*)alloc((size_t)B * H * 4);
  float*     esc    = (float*)alloc((size_t)E_BP * 4);
  float*     emax   = (float*)alloc((size_t)N_PROT * 4);
  float*     denom  = (float*)alloc((size_t)N_PROT * 4);
  float*     sums   = (float*)alloc(2 * 256 * 4);
  _Float16*  wina16 = (_Float16*)alloc(32 * H * 2);
  _Float16*  winp16 = (_Float16*)alloc((size_t)F_PROT * H * 2);
  _Float16*  wl1bp  = (_Float16*)alloc(H * H * 2);
  _Float16*  wr1bp  = (_Float16*)alloc(H * H * 2);
  _Float16*  wl1pp  = (_Float16*)alloc(H * H * 2);
  _Float16*  wr1pp  = (_Float16*)alloc(H * H * 2);
  _Float16*  wl2bp  = (_Float16*)alloc(H * H * 2);
  _Float16*  wr2bp  = (_Float16*)alloc(H * H * 2);
  _Float16*  wl2pp  = (_Float16*)alloc(H * H * 2);
  _Float16*  wr2pp  = (_Float16*)alloc(H * H * 2);
  _Float16*  wpo16  = (_Float16*)alloc(2 * H * H * 2);
  _Float16*  wou16  = (_Float16*)alloc((size_t)H * V * 2);

  // --- launch helpers
  auto pack = [&](const float* W, _Float16* Wf, int K, int N, int nKs) {
    int n = N * nKs * 32;
    packB_kernel<<<(n + 255) / 256, 256, 0, stream>>>(W, Wf, K, N, nKs);
  };
  auto gemmF = [&](const float* A, int lda, const _Float16* Wf, int nKs, int N,
                   const float* bias, float* C, int ldc, int M) {
    dim3 g(N / 16, (M / 16 + 3) / 4);
    gemm_wmma<float><<<g, 128, 0, stream>>>(A, lda, (const v16h*)Wf, nKs, bias, C, ldc, M);
  };
  auto gemmH = [&](const _Float16* A, int lda, const _Float16* Wf, int nKs, int N,
                   const float* bias, float* C, int ldc, int M) {
    dim3 g(N / 16, (M / 16 + 3) / 4);
    gemm_wmma<_Float16><<<g, 128, 0, stream>>>(A, lda, (const v16h*)Wf, nKs, bias, C, ldc, M);
  };
  auto bnPR = [&](float* x, const float* aP, int aScalar, const float* g, const float* b,
                  int Nr, int C) {
    hipMemsetAsync(sums, 0, 2 * C * sizeof(float), stream);
    bn_stats_prelu_kernel<<<imin_(Nr, 512), C, 0, stream>>>(x, aP, aScalar, sums, Nr, C);
    int n = Nr * C;
    bn_apply_prelu_kernel<float><<<(n + 255) / 256, 256, 0, stream>>>(x, aP, aScalar,
                                                                      sums, g, b, x, Nr, C);
  };
  auto run_edges = [&](const int* src, const int* dst, int E,
                       const float* xl, const float* xr, const float* att, float* out) {
    fill_kernel<<<(N_PROT + 255) / 256, 256, 0, stream>>>(emax, -INFINITY, N_PROT);
    hipMemsetAsync(denom, 0, N_PROT * sizeof(float), stream);
    int gb = (E + 7) / 8;  // 8 waves / 256-thread block, one wave per edge
    edge_score_kernel<<<gb, 256, 0, stream>>>(src, dst, xl, xr, att, esc, emax, E);
    edge_exp_kernel<<<(E + 255) / 256, 256, 0, stream>>>(dst, esc, emax, denom, E);
    edge_scatter_kernel<<<gb, 256, 0, stream>>>(src, dst, esc, denom, xl, out, E);
  };

  // --- pack all weights into f16 WMMA fragments (padded K)
  pack(Win_aa, wina16, F_AA, H, 1);
  pack(Win_prot, winp16, F_PROT, H, F_PROT / 32);
  pack(c1bp.Wl, wl1bp, H, H, 4);  pack(c1bp.Wr, wr1bp, H, H, 4);
  pack(c1pp.Wl, wl1pp, H, H, 4);  pack(c1pp.Wr, wr1pp, H, H, 4);
  pack(c2bp.Wl, wl2bp, H, H, 4);  pack(c2bp.Wr, wr2bp, H, H, 4);
  pack(c2pp.Wl, wl2pp, H, H, 4);  pack(c2pp.Wr, wr2pp, H, H, 4);
  pack(Wpost, wpo16, 2 * H, H, 8);
  pack(Wout, wou16, H, V, 4);

  // --- stage A: per-type lin_in + PReLU + BN
  {
    int n = N_AA * 32;
    padA_kernel<<<(n + 255) / 256, 256, 0, stream>>>(x_aa, xa16, N_AA, F_AA, 32);
  }
  gemmH(xa16, 32, wina16, 1, H, bin_aa, yaa, H, N_AA);
  {
    int n = N_AA * H;
    hipMemsetAsync(sums, 0, 2 * H * sizeof(float), stream);
    bn_stats_prelu_kernel<<<512, H, 0, stream>>>(yaa, a1_aa, 0, sums, N_AA, H);
    bn_apply_prelu_kernel<_Float16><<<(n + 255) / 256, 256, 0, stream>>>(
        yaa, a1_aa, 0, sums, g1_aa, b1_aa, h_aa, N_AA, H);
  }
  gemmF(x_prot, F_PROT, winp16, F_PROT / 32, H, bin_prot, h_pr, H, N_PROT);
  bnPR(h_pr, a1_prot, 0, g1_prot, b1_prot, N_PROT, H);

  float* xlbig = yaa;  // 500k x 128 f32 buffer now free -> reuse for GAT xl

  // --- HeteroConv layer 1 -> g1
  hipMemsetAsync(g1, 0, (size_t)N_PROT * H * sizeof(float), stream);
  gemmH(h_aa, H, wl1bp, 4, H, c1bp.bl, xlbig, H, N_AA);
  gemmF(h_pr, H, wr1bp, 4, H, c1bp.br, smallB, H, N_PROT);
  run_edges(ei_bp, ei_bp + E_BP, E_BP, xlbig, smallB, c1bp.att, g1);
  gemmF(h_pr, H, wl1pp, 4, H, c1pp.bl, smallA, H, N_PROT);
  gemmF(h_pr, H, wr1pp, 4, H, c1pp.br, smallB, H, N_PROT);
  run_edges(ei_pp, ei_pp + E_PP, E_PP, smallA, smallB, c1pp.att, g1);
  {
    int n = N_PROT * H;
    add2bias_kernel<<<(n + 255) / 256, 256, 0, stream>>>(g1, c1bp.bias, c1pp.bias, n, H);
    bnPR(g1, a_g1, 1, g_g1, b_g1, N_PROT, H);
  }

  // --- HeteroConv layer 2 -> g2 (aa side reuses h_aa)
  hipMemsetAsync(g2, 0, (size_t)N_PROT * H * sizeof(float), stream);
  gemmH(h_aa, H, wl2bp, 4, H, c2bp.bl, xlbig, H, N_AA);
  gemmF(g1, H, wr2bp, 4, H, c2bp.br, smallB, H, N_PROT);
  run_edges(ei_bp, ei_bp + E_BP, E_BP, xlbig, smallB, c2bp.att, g2);
  gemmF(g1, H, wl2pp, 4, H, c2pp.bl, smallA, H, N_PROT);
  gemmF(g1, H, wr2pp, 4, H, c2pp.br, smallB, H, N_PROT);
  run_edges(ei_pp, ei_pp + E_PP, E_PP, smallA, smallB, c2pp.att, g2);
  {
    int n = N_PROT * H;
    add2bias_kernel<<<(n + 255) / 256, 256, 0, stream>>>(g2, c2bp.bias, c2pp.bias, n, H);
    bnPR(g2, a_g2, 1, g_g2, b_g2, N_PROT, H);
  }

  // --- readout head on seed proteins
  {
    int n = B * 2 * H;
    concat_kernel<<<(n + 255) / 256, 256, 0, stream>>>(h_pr, g2, z, B, H);
    bnPR(z, a_pr, 1, g_pr, b_pr, B, 2 * H);
  }
  gemmF(z, 2 * H, wpo16, 8, H, bpost, z2, H, B);
  bnPR(z2, a_post, 1, g_post, b_post, B, H);
  gemmF(z2, H, wou16, 4, V, bout, (float*)d_out, V, B);
}
